// BasicLSTM_71047349010480
// MI455X (gfx1250) — matrix operand order repaired and checked by reading, not compiled, not run on hardware
//
#include <hip/hip_runtime.h>
#include <cstdint>
#include <cstddef>

// Problem dims (fixed by reference)
#define T_STEPS 8192
#define IN_DIM  1024
#define H_DIM   1024
#define G_DIM   4096   // 4*H
#define CMB     2048   // IN+H (W_w row pitch)

#define RBLK 256       // persistent recurrent blocks
#define RTHR 256       // threads per recurrent block

typedef __attribute__((ext_vector_type(16))) __bf16 v16bf;
typedef __attribute__((ext_vector_type(8)))  float  v8f;

__device__ __forceinline__ unsigned short f2bf(float f) {
  unsigned u = __float_as_uint(f);
  u += 0x7fffu + ((u >> 16) & 1u);          // round-to-nearest-even
  return (unsigned short)(u >> 16);
}
__device__ __forceinline__ float sigmoidf_(float x) {
  return 1.0f / (1.0f + __expf(-x));
}

// ---------------------------------------------------------------------------
// f32 -> bf16 conversion (grid-stride)
// ---------------------------------------------------------------------------
__global__ void cvt_f32_bf16(const float* __restrict__ in,
                             unsigned short* __restrict__ out, int n) {
  int i = blockIdx.x * blockDim.x + threadIdx.x;
  int stride = gridDim.x * blockDim.x;
  for (; i < n; i += stride) out[i] = f2bf(in[i]);
}

// ---------------------------------------------------------------------------
// C[M,N](f32) = A[M,K](bf16, row-major, lda) @ W[N,K](bf16, row-major, ldw,
//               col offset koff)^T + bias[N]
// One wave per 32x32 output tile: 2x2 register-blocked
// v_wmma_f32_16x16x32_bf16 (4 WMMAs per K-step, A/B frags each reused 2x).
// Fragment layouts per CDNA5 ISA 7.12.2.
// ---------------------------------------------------------------------------
union Frag32B { v16bf v; uint4 q[2]; };

__global__ __launch_bounds__(256)
void gemm_bf16_wmma(const unsigned short* __restrict__ A, int lda,
                    const unsigned short* __restrict__ W, int ldw, int koff,
                    const float* __restrict__ bias,
                    float* __restrict__ C, int ldc,
                    int M, int N, int K) {
  const int lane = threadIdx.x & 31;
  const int wid  = threadIdx.x >> 5;
  const int tiles_n = N >> 5;               // 32-wide super-tiles
  const int gw = blockIdx.x * 8 + wid;
  const int tm = gw / tiles_n;
  const int tn = gw % tiles_n;
  if (tm * 32 >= M) return;                 // grid sized exactly; uniform guard

  const int m0 = tm * 32, n0 = tn * 32;
  const int hl = lane & 15;                 // row within tile (A) / col (B,C)
  const int kg = lane >> 4;                 // K half-group select

  const unsigned short* arow0 = A + (size_t)(m0 + hl) * lda;
  const unsigned short* arow1 = arow0 + (size_t)16 * lda;
  const unsigned short* wrow0 = W + (size_t)(n0 + hl) * ldw + koff;
  const unsigned short* wrow1 = wrow0 + (size_t)16 * ldw;

  v8f acc00 = {}, acc01 = {}, acc10 = {}, acc11 = {};
  for (int k0 = 0; k0 < K; k0 += 32) {
    Frag32B a0, a1, b0, b1;
    // A 16x32 bf16: lane holds K = {k0+8*kg .. +7} and {k0+16+8*kg .. +7}
    a0.q[0] = *(const uint4*)(arow0 + k0 + kg * 8);
    a0.q[1] = *(const uint4*)(arow0 + k0 + 16 + kg * 8);
    a1.q[0] = *(const uint4*)(arow1 + k0 + kg * 8);
    a1.q[1] = *(const uint4*)(arow1 + k0 + 16 + kg * 8);
    // B 32x16 bf16 (= W^T): lane holds K = {k0+8*kg .. +7} and {k0+16+8*kg .. +7}
    const unsigned short* bp0 = wrow0 + k0 + kg * 8;
    const unsigned short* bp1 = wrow1 + k0 + kg * 8;
    b0.q[0] = *(const uint4*)(bp0);
    b0.q[1] = *(const uint4*)(bp0 + 16);
    b1.q[0] = *(const uint4*)(bp1);
    b1.q[1] = *(const uint4*)(bp1 + 16);

    acc00 = __builtin_amdgcn_wmma_f32_16x16x32_bf16(false, a0.v, false, b0.v,
                                                    (short)0, acc00, false, false);
    acc01 = __builtin_amdgcn_wmma_f32_16x16x32_bf16(false, a0.v, false, b1.v,
                                                    (short)0, acc01, false, false);
    acc10 = __builtin_amdgcn_wmma_f32_16x16x32_bf16(false, a1.v, false, b0.v,
                                                    (short)0, acc10, false, false);
    acc11 = __builtin_amdgcn_wmma_f32_16x16x32_bf16(false, a1.v, false, b1.v,
                                                    (short)0, acc11, false, false);
  }

  // C/D layout: lane -> col n0+hl (+16 for right blocks); VGPR i -> row
  // m0 + 8*kg + i (+16 for lower blocks).
  const float bv0 = bias ? bias[n0 + hl] : 0.0f;
  const float bv1 = bias ? bias[n0 + 16 + hl] : 0.0f;
  float* c00 = C + (size_t)(m0 + kg * 8) * ldc + (n0 + hl);
  float* c10 = c00 + (size_t)16 * ldc;
#pragma unroll
  for (int i = 0; i < 8; ++i) {
    c00[(size_t)i * ldc]      = acc00[i] + bv0;
    c00[(size_t)i * ldc + 16] = acc01[i] + bv1;
    c10[(size_t)i * ldc]      = acc10[i] + bv0;
    c10[(size_t)i * ldc + 16] = acc11[i] + bv1;
  }
}

// ---------------------------------------------------------------------------
// Persistent recurrent kernel.
//   256 blocks x 256 threads; block b owns hidden units [4b, 4b+4) => 16 gate
//   rows (i,f,g,o). Thread (r=tid>>4, j=tid&15) owns 64 bf16 recurrent weights
//   of row r, k-range [64j, 64j+64) -- held in 32 VGPRs for all 8192 steps.
//   Per step: h (4KB, double-buffered in ws) -> LDS; 64 FMAs/thread;
//   16-lane shuffle reduction; LSTM cell update for 4 units; global barrier.
// ---------------------------------------------------------------------------
__global__ __launch_bounds__(RTHR)
void lstm_rec(const unsigned short* __restrict__ Wbf,   // [4096 x 2048] bf16
              const float* __restrict__ Xp,             // [T x 4096] f32
              float* __restrict__ hpool,                // [2 x 1024] f32 (zeroed)
              unsigned short* __restrict__ Hbuf,        // [T x 1024] bf16 out
              unsigned* __restrict__ bar) {             // zeroed counter
  __shared__ float h_lds[H_DIM];
  __shared__ float gv[16];
  __shared__ float cst[4];

  const int tid = threadIdx.x;
  const int r = tid >> 4;                  // local gate row 0..15
  const int j = tid & 15;                  // k-slice 0..15
  const int ub = blockIdx.x * 4;           // first hidden unit of this block
  const int g = r >> 2, u = r & 3;
  const int grow = g * H_DIM + ub + u;     // global gate row in [0,4096)

  // Preload recurrent weights (W_w columns IN..IN+H) into registers.
  unsigned int WR[32];
  const unsigned int* wp =
      (const unsigned int*)(Wbf + (size_t)grow * CMB + IN_DIM + j * 64);
#pragma unroll
  for (int k = 0; k < 32; ++k) WR[k] = wp[k];

  if (tid < 4) cst[tid] = 0.0f;
  __syncthreads();

  for (int t = 0; t < T_STEPS; ++t) {
    // Broadcast current h into LDS (volatile: bypass stale near caches).
    {
      const volatile float* hsrc = hpool + (size_t)(t & 1) * H_DIM;
#pragma unroll
      for (int i = 0; i < H_DIM / RTHR; ++i)
        h_lds[tid + i * RTHR] = hsrc[tid + i * RTHR];
    }
    __syncthreads();

    // Partial dot: 64 weights (32 packed bf16 pairs) against h slice.
    float sum = 0.0f;
    const float* hp = &h_lds[j * 64];
#pragma unroll
    for (int k = 0; k < 32; ++k) {
      const unsigned int w = WR[k];
      const float w0 = __uint_as_float(w << 16);
      const float w1 = __uint_as_float(w & 0xffff0000u);
      sum = fmaf(w0, hp[2 * k], sum);
      sum = fmaf(w1, hp[2 * k + 1], sum);
    }
    // Reduce 16 lanes (rows occupy 16-lane halves of each wave32).
    sum += __shfl_xor(sum, 8, 32);
    sum += __shfl_xor(sum, 4, 32);
    sum += __shfl_xor(sum, 2, 32);
    sum += __shfl_xor(sum, 1, 32);
    if (j == 0) gv[r] = sum + Xp[(size_t)t * G_DIM + grow];
    __syncthreads();

    // LSTM cell update for the 4 owned units.
    if (tid < 4) {
      const float iv = gv[tid];
      const float fv = gv[4 + tid];
      const float gg = gv[8 + tid];
      const float ov = gv[12 + tid];
      const float c  = sigmoidf_(fv) * cst[tid] + sigmoidf_(iv) * tanhf(gg);
      const float h  = sigmoidf_(ov) * tanhf(c);
      cst[tid] = c;
      hpool[(size_t)((t + 1) & 1) * H_DIM + ub + tid] = h;
      Hbuf[(size_t)t * H_DIM + ub + tid] = f2bf(h);
    }
    __threadfence();

    // Grid-wide monotonic barrier (phase = t): max count 8192*256 < 2^32.
    __syncthreads();
    if (tid == 0) {
      atomicAdd(bar, 1u);
      const unsigned target = (unsigned)(t + 1) * RBLK;
      while (atomicAdd(bar, 0u) < target) __builtin_amdgcn_s_sleep(1);
    }
    __syncthreads();
  }
}

// ---------------------------------------------------------------------------
extern "C" void kernel_launch(void* const* d_in, const int* in_sizes, int n_in,
                              void* d_out, int out_size, void* d_ws, size_t ws_size,
                              hipStream_t stream) {
  (void)in_sizes; (void)n_in; (void)out_size; (void)ws_size;
  const float* x     = (const float*)d_in[0];  // [T,1,IN]
  const float* W_w   = (const float*)d_in[1];  // [4H, IN+H]
  const float* W_b   = (const float*)d_in[2];  // [4H]
  const float* out_w = (const float*)d_in[3];  // [OUT, H]
  const float* out_b = (const float*)d_in[4];  // [OUT]
  float* out = (float*)d_out;                  // [T,1,OUT] f32

  char* ws = (char*)d_ws;
  size_t o = 0;
  auto alloc = [&](size_t b) { size_t r = o; o = (o + b + 255) & ~(size_t)255; return r; };
  unsigned*       bar   = (unsigned*)      (ws + alloc(256));
  float*          hpool = (float*)         (ws + alloc((size_t)2 * H_DIM * 4));
  unsigned short* xbf   = (unsigned short*)(ws + alloc((size_t)T_STEPS * IN_DIM * 2));
  unsigned short* wbf   = (unsigned short*)(ws + alloc((size_t)G_DIM * CMB * 2));
  unsigned short* owbf  = (unsigned short*)(ws + alloc((size_t)H_DIM * H_DIM * 2));
  unsigned short* hbuf  = (unsigned short*)(ws + alloc((size_t)T_STEPS * H_DIM * 2));
  float*          xproj = (float*)         (ws + alloc((size_t)T_STEPS * G_DIM * 4));

  // Zero barrier counter + h double-buffer (deterministic per launch/replay).
  hipMemsetAsync(ws, 0, 16384, stream);

  // f32 -> bf16 conversions
  cvt_f32_bf16<<<2048, 256, 0, stream>>>(x, xbf, T_STEPS * IN_DIM);
  cvt_f32_bf16<<<2048, 256, 0, stream>>>(W_w, wbf, G_DIM * CMB);
  cvt_f32_bf16<<<1024, 256, 0, stream>>>(out_w, owbf, H_DIM * H_DIM);

  // Phase 1: Xproj[T,4H] = x @ W_x^T + W_b   (WMMA bf16, 32x32/wave)
  {
    const int waves = (T_STEPS / 32) * (G_DIM / 32);
    gemm_bf16_wmma<<<waves / 8, 256, 0, stream>>>(
        xbf, IN_DIM, wbf, CMB, /*koff=*/0, W_b, xproj, G_DIM,
        T_STEPS, G_DIM, IN_DIM);
  }

  // Phase 2: persistent recurrence (register-resident W_h, grid barrier/step)
  lstm_rec<<<RBLK, RTHR, 0, stream>>>(wbf, xproj, hpool, hbuf, bar);

  // Phase 3: Y[T,OUT] = H @ out_w^T + out_b  (WMMA bf16, 32x32/wave)
  {
    const int waves = (T_STEPS / 32) * (H_DIM / 32);
    gemm_bf16_wmma<<<waves / 8, 256, 0, stream>>>(
        hbuf, H_DIM, owbf, H_DIM, /*koff=*/0, out_b, out, H_DIM,
        T_STEPS, H_DIM, H_DIM);
  }
}
